// _LightGCN_25632364822918
// MI455X (gfx1250) — matrix operand order repaired and checked by reading, not compiled, not run on hardware
//
#include <hip/hip_runtime.h>
#include <hip/hip_bf16.h>

#define NUQ 100000
#define NIQ 50000
#define NTOT (NUQ + NIQ)
#define CUQ 100
#define CIQ 100
#define CNQ (CUQ + CIQ)
#define DQ 64
#define NLQ 3

typedef __attribute__((ext_vector_type(16))) _Float16 v16h;
typedef __attribute__((ext_vector_type(8)))  float    v8f;
typedef __attribute__((ext_vector_type(4)))  unsigned int v4u;
typedef __attribute__((ext_vector_type(8)))  int      v8i;
typedef __attribute__((ext_vector_type(4)))  int      v4i;

// ---------------- elementwise helpers ----------------

__global__ void fill_zero_kernel(float* __restrict__ p, long long n) {
  long long i = (long long)blockIdx.x * blockDim.x + threadIdx.x;
  if (i < n) p[i] = 0.0f;
}

__global__ void copy_kernel(float* __restrict__ dst, const float* __restrict__ src, long long n) {
  long long i = (long long)blockIdx.x * blockDim.x + threadIdx.x;
  if (i < n) dst[i] = src[i];
}

__global__ void axpy_kernel(float* __restrict__ acc, const float* __restrict__ x, long long n) {
  long long i = (long long)blockIdx.x * blockDim.x + threadIdx.x;
  if (i < n) acc[i] += x[i];
}

__global__ void scale_kernel(float* __restrict__ p, float s, long long n) {
  long long i = (long long)blockIdx.x * blockDim.x + threadIdx.x;
  if (i < n) p[i] *= s;
}

// ---------------- SpMM: y[rows[e]] += vals[e] * x[cols[e]] over 64 dims ----------------
// 32 lanes per edge, each lane handles 2 consecutive floats (one float2 gather, two atomics).
// x (38.4 MB) lives in the 192 MB L2, so the random gathers are L2 hits after first touch.

__global__ void spmm_kernel(const int* __restrict__ rows, const int* __restrict__ cols,
                            const float* __restrict__ vals, const float* __restrict__ x,
                            float* __restrict__ y, int nnz) {
  long long t = (long long)blockIdx.x * blockDim.x + threadIdx.x;
  int e    = (int)(t >> 5);
  int lane = (int)(t & 31);
  if (e >= nnz) return;
  // stream-ahead hint on the edge arrays (emits global_prefetch_b8)
  if (lane == 0 && e + 2048 < nnz) {
    __builtin_prefetch(&rows[e + 2048], 0, 1);
    __builtin_prefetch(&cols[e + 2048], 0, 1);
  }
  int   r = rows[e];
  int   c = cols[e];
  float v = vals[e];
  const float2* xr = (const float2*)(x + (long long)c * DQ);
  float2 xv = xr[lane];
  float* yr = y + (long long)r * DQ + lane * 2;
  atomicAdd(yr,     v * xv.x);
  atomicAdd(yr + 1, v * xv.y);
}

// ---------------- Projection: out[c][d] += sum_u W[u][c] * E[u][d]  (ncols == 100) ----------------

__global__ void proj_kernel(const float* __restrict__ W, const float* __restrict__ E,
                            float* __restrict__ out, int nrows, int ncols) {
  __shared__ float acc[CUQ][DQ];   // 25.6 KB
  int t = threadIdx.x;             // 256 threads
  for (int i = t; i < CUQ * DQ; i += 256) ((float*)acc)[i] = 0.0f;
  __syncthreads();
  int d  = t & 63;
  int cg = t >> 6;                 // 0..3
  int u0 = blockIdx.x * 512;
  int u1 = u0 + 512; if (u1 > nrows) u1 = nrows;
  for (int u = u0; u < u1; ++u) {
    float ev = E[(long long)u * DQ + d];
    const float* wr = W + (long long)u * ncols;
    for (int c = cg; c < ncols; c += 4)
      acc[c][d] += wr[c] * ev;     // unique (c,d) per thread -> no LDS races
  }
  __syncthreads();
  for (int i = t; i < ncols * DQ; i += 256)
    atomicAdd(&out[i], ((float*)acc)[i]);
}

// ---------------- WMMA lift GEMM: Out[M][64] = A[M][100] * Bm[100][64] ----------------
// A tile (32x100 f32) staged into LDS by the Tensor Data Mover (wave 0 issues one
// tensor_load_to_lds; TDM OOB-zero handles the M tail via tensor_dim1 = M - m0).
// K padded to 128; f32 -> f16 in LDS; 8 waves per block cover a 32x64 tile.

__global__ __launch_bounds__(256)
void lift_wmma_kernel(const float* __restrict__ A, const float* __restrict__ Bm,
                      float* __restrict__ Out, int M) {
  const int K  = CUQ;   // 100
  const int KP = 128;
  __shared__ float    Astage[32][CUQ]; // 12.8 KB, fp32 tile staged by TDM
  __shared__ _Float16 As[32][KP];      // 8 KB
  __shared__ _Float16 Bs[DQ][KP];      // 16 KB, transposed: Bs[n][k]
  int t  = threadIdx.x;
  int m0 = blockIdx.x * 32;

#if __has_builtin(__builtin_amdgcn_tensor_load_to_lds) && defined(__gfx1250__)
  if (t < 32) {  // wave 0 issues the TDM op (EXEC-independent, per-wave TENSORcnt)
    unsigned long long ga = (unsigned long long)(A + (long long)m0 * K);
    // compile-probe environment: use low 32 bits of the flat LDS address as lds_addr
    unsigned lds_off = (unsigned)(unsigned long long)(void*)&Astage[0][0];
    v4u g0 = {0u, 0u, 0u, 0u};
    g0[0] = 1u;                                        // count=1, user descriptor
    g0[1] = lds_off;                                   // lds_addr
    g0[2] = (unsigned)(ga & 0xffffffffull);            // global_addr[31:0]
    g0[3] = (unsigned)((ga >> 32) & 0x1ffffffu) | (2u << 30);  // addr[56:32] | type=2
    int rem = M - m0;                                  // tensor_dim1: OOB rows read as 0
    v8i g1 = {0, 0, 0, 0, 0, 0, 0, 0};
    g1[0] = (int)(2u << 16);                           // data_size=4B (code 2)
    g1[1] = (int)((unsigned)K << 16);                  // tensor_dim0[15:0] @ bit48
    g1[2] = (int)((unsigned)rem << 16);                // tensor_dim1[15:0] @ bit80
    g1[3] = (int)((unsigned)K << 16);                  // tile_dim0 @ bit112
    g1[4] = 32;                                        // tile_dim1 @ bit128
    g1[5] = K;                                         // tensor_dim0_stride @ bit160
    v4i g2 = {0, 0, 0, 0};
    v4i g3 = {0, 0, 0, 0};
    v8i g4 = {0, 0, 0, 0, 0, 0, 0, 0};                 // trailing group (clang-23 6-arg form)
    __builtin_amdgcn_tensor_load_to_lds(g0, g1, g2, g3, g4, 0);
    __builtin_amdgcn_s_wait_tensorcnt(0);
  }
  // overlap: all threads stage/convert B while the TDM fills Astage
  for (int i = t; i < DQ * KP; i += 256) {
    int n = i >> 7;
    int k = i & 127;
    Bs[n][k] = (k < K) ? (_Float16)Bm[k * DQ + n] : (_Float16)0.0f;
  }
  __syncthreads();
  for (int i = t; i < 32 * KP; i += 256) {
    int r = i >> 7;
    int k = i & 127;
    As[r][k] = (k < K) ? (_Float16)Astage[r][k] : (_Float16)0.0f;
  }
#else
  for (int i = t; i < DQ * KP; i += 256) {
    int n = i >> 7;
    int k = i & 127;
    Bs[n][k] = (k < K) ? (_Float16)Bm[k * DQ + n] : (_Float16)0.0f;
  }
  for (int i = t; i < 32 * KP; i += 256) {
    int r  = i >> 7;
    int k  = i & 127;
    int gr = m0 + r;
    As[r][k] = (k < K && gr < M) ? (_Float16)A[(long long)gr * K + k] : (_Float16)0.0f;
  }
#endif
  __syncthreads();

  int w    = t >> 5;          // wave 0..7
  int lane = t & 31;
  int mt   = (w >> 2) * 16;   // 0 or 16
  int nt   = (w & 3) * 16;    // 0,16,32,48
  int lh   = lane >> 4;       // lane half
  int ll   = lane & 15;

  int arow = mt + ll;
  int bcol = nt + ll;

  v8f c = {};
  #pragma unroll
  for (int kk = 0; kk < KP; kk += 32) {
    v16h a, b;
    #pragma unroll
    for (int v = 0; v < 8; ++v) {
      // 16-bit A-matrix 16x32 layout: VGPR 0..3 -> K = lh*8 + 2v{,+1}; VGPR 4..7 -> +16
      int ka = kk + ((v < 4) ? (lh * 8 + 2 * v) : (16 + lh * 8 + 2 * (v - 4)));
      a[2 * v]     = As[arow][ka];
      a[2 * v + 1] = As[arow][ka + 1];
      // B 32x16: same K order as A: VGPR 0..3 -> K = lh*8 + 2v{,+1}; VGPR 4..7 -> +16
      int kb = kk + ((v < 4) ? (lh * 8 + 2 * v) : (16 + lh * 8 + 2 * (v - 4)));
      b[2 * v]     = Bs[bcol][kb];
      b[2 * v + 1] = Bs[bcol][kb + 1];
    }
    c = __builtin_amdgcn_wmma_f32_16x16x32_f16(false, a, false, b, (short)0, c, false, false);
  }

  // C/D layout: lanes 0-15 hold M=r, lanes 16-31 hold M=8+r
  int rbase = m0 + mt + lh * 8;
  int gcol  = nt + ll;
  if (m0 + 32 <= M) {          // full-tile fast path: straight-line stores
    #pragma unroll
    for (int r = 0; r < 8; ++r)
      Out[(long long)(rbase + r) * DQ + gcol] = c[r];
  } else {
    #pragma unroll
    for (int r = 0; r < 8; ++r)
      if (rbase + r < M) Out[(long long)(rbase + r) * DQ + gcol] = c[r];
  }
}

// ---------------- final BPR dots: out[b] = <u, i_pos - i_neg> over 4*64 features ----------------

__global__ void bpr_kernel(const float* __restrict__ e0, const float* __restrict__ e1,
                           const float* __restrict__ e2, const float* __restrict__ u3,
                           const float* __restrict__ i3,
                           const int* __restrict__ users, const int* __restrict__ items,
                           const int* __restrict__ negs, float* __restrict__ out, int B) {
  __shared__ float red[DQ];
  int b = blockIdx.x;
  if (b >= B) return;
  int d  = threadIdx.x;  // 64 threads
  int u  = users[b];
  int ip = items[b];
  int in_ = negs[b];
  long long uo = (long long)u * DQ + d;
  long long po = (long long)(NUQ + ip) * DQ + d;
  long long no = (long long)(NUQ + in_) * DQ + d;
  float s = 0.0f;
  s += e0[uo] * (e0[po] - e0[no]);
  s += e1[uo] * (e1[po] - e1[no]);
  s += e2[uo] * (e2[po] - e2[no]);
  s += u3[(long long)u * DQ + d] * (i3[(long long)ip * DQ + d] - i3[(long long)in_ * DQ + d]);
  red[d] = s;
  __syncthreads();
  for (int off = 32; off > 0; off >>= 1) {
    if (d < off) red[d] += red[d + off];
    __syncthreads();
  }
  if (d == 0) out[b] = red[0];
}

// ---------------- host orchestration ----------------

extern "C" void kernel_launch(void* const* d_in, const int* in_sizes, int n_in,
                              void* d_out, int out_size, void* d_ws, size_t ws_size,
                              hipStream_t stream) {
  const float* user_emb = (const float*)d_in[0];
  const float* item_emb = (const float*)d_in[1];
  const int*   gr[3] = {(const int*)d_in[2], (const int*)d_in[5], (const int*)d_in[8]};
  const int*   gc[3] = {(const int*)d_in[3], (const int*)d_in[6], (const int*)d_in[9]};
  const float* gv[3] = {(const float*)d_in[4], (const float*)d_in[7], (const float*)d_in[10]};
  const float* uc    = (const float*)d_in[11];
  const float* ic    = (const float*)d_in[12];
  const int*   cgr   = (const int*)d_in[13];
  const int*   cgc   = (const int*)d_in[14];
  const float* cgv   = (const float*)d_in[15];
  const int*   users = (const int*)d_in[16];
  const int*   items = (const int*)d_in[17];
  const int*   negs  = (const int*)d_in[18];
  float* out = (float*)d_out;

  int nnz  = in_sizes[2];
  int cnnz = in_sizes[13];
  int B    = in_sizes[16];

  const long long nd = (long long)NTOT * DQ;
  float* ws = (float*)d_ws;
  float* x      = ws; ws += nd;
  float* y      = ws; ws += nd;
  float* acc[3];
  for (int g = 0; g < 3; ++g) { acc[g] = ws; ws += nd; }
  float* ecin   = ws; ws += (long long)CNQ * DQ;
  float* ecx    = ws; ws += (long long)CNQ * DQ;
  float* ecy    = ws; ws += (long long)CNQ * DQ;
  float* ecacc  = ws; ws += (long long)CNQ * DQ;
  float* u3     = ws; ws += (long long)NUQ * DQ;
  float* i3     = ws; ws += (long long)NIQ * DQ;

  auto blks = [](long long n) { return (unsigned)((n + 255) / 256); };
  auto zero = [&](float* p, long long n) {
    fill_zero_kernel<<<blks(n), 256, 0, stream>>>(p, n);
  };
  auto copy = [&](float* d, const float* s, long long n) {
    copy_kernel<<<blks(n), 256, 0, stream>>>(d, s, n);
  };

  // ---- three GCN passes on the big graph(s) ----
  for (int g = 0; g < 3; ++g) {
    copy(x, user_emb, (long long)NUQ * DQ);
    copy(x + (long long)NUQ * DQ, item_emb, (long long)NIQ * DQ);
    copy(acc[g], x, nd);
    float* cx = x;
    float* cy = y;
    unsigned sblk = blks((long long)nnz * 32);
    for (int l = 0; l < NLQ; ++l) {
      zero(cy, nd);
      spmm_kernel<<<sblk, 256, 0, stream>>>(gr[g], gc[g], gv[g], cx, cy, nnz);
      axpy_kernel<<<blks(nd), 256, 0, stream>>>(acc[g], cy, nd);
      float* tmp = cx; cx = cy; cy = tmp;
    }
    scale_kernel<<<blks(nd), 256, 0, stream>>>(acc[g], 0.25f, nd);
  }

  // ---- community projection + tiny GCN ----
  const long long cnd = (long long)CNQ * DQ;
  zero(ecin, cnd);
  proj_kernel<<<(NUQ + 511) / 512, 256, 0, stream>>>(uc, user_emb, ecin, NUQ, CUQ);
  proj_kernel<<<(NIQ + 511) / 512, 256, 0, stream>>>(ic, item_emb, ecin + (long long)CUQ * DQ, NIQ, CIQ);
  copy(ecx, ecin, cnd);
  copy(ecacc, ecin, cnd);
  {
    float* cx = ecx;
    float* cy = ecy;
    unsigned sblk = blks((long long)cnnz * 32);
    for (int l = 0; l < NLQ; ++l) {
      zero(cy, cnd);
      spmm_kernel<<<sblk, 256, 0, stream>>>(cgr, cgc, cgv, cx, cy, cnnz);
      axpy_kernel<<<blks(cnd), 256, 0, stream>>>(ecacc, cy, cnd);
      float* tmp = cx; cx = cy; cy = tmp;
    }
    scale_kernel<<<blks(cnd), 256, 0, stream>>>(ecacc, 0.25f, cnd);
  }

  // ---- WMMA lift-back GEMMs (TDM-staged A tiles) ----
  lift_wmma_kernel<<<(NUQ + 31) / 32, 256, 0, stream>>>(uc, ecacc, u3, NUQ);
  lift_wmma_kernel<<<(NIQ + 31) / 32, 256, 0, stream>>>(ic, ecacc + (long long)CUQ * DQ, i3, NIQ);

  // ---- final dots ----
  bpr_kernel<<<B, DQ, 0, stream>>>(acc[0], acc[1], acc[2], u3, i3, users, items, negs, out, B);
}
